// PUGNNRadius_77601469104647
// MI455X (gfx1250) — matrix operand order repaired and checked by reading, not compiled, not run on hardware
//
#include <hip/hip_runtime.h>
#include <hip/hip_bf16.h>

// ---------------------------------------------------------------------------
// PU-GNN (radius GraphConv x3 -> kNN NodeShuffle -> MLP) for gfx1250.
// Dense GEMMs: V_WMMA_F32_16X16X32_BF16 with bf16x3 error compensation
// (~fp32 accuracy), 16x64 tile per wave (A-fragment reuse across 4 N-tiles),
// dual-input fused GEMM for GraphConv (root+rel in one pass).
// Graph aggregations: exact fp32 sparse gathers (radius graph ~25-50 nbrs,
// kNN graph exactly 16 -> dense masked GEMM would waste 17 GFLOP + 64 MB).
// ---------------------------------------------------------------------------

#define NPTS   4096
#define CCH    256
#define RUP    4
#define KNN    16
#define R2     (0.4f * 0.4f)

typedef __attribute__((ext_vector_type(16))) __bf16 v16bf;
typedef __attribute__((ext_vector_type(8)))  float  v8f;

__device__ __forceinline__ void split_bf(float f, __bf16& hi, __bf16& lo) {
    // hi = truncate-to-bf16 (exactly representable), lo = RNE(residual)
    float fh = __uint_as_float(__float_as_uint(f) & 0xffff0000u);
    hi = (__bf16)fh;
    lo = (__bf16)(f - fh);
}

// Load + split an A fragment (16x32; contiguous k per lane half, same order as B):
// lane holds row ml; k(e) = kk + (lh?16:0) + e
__device__ __forceinline__ void load_a_frag(const float* __restrict__ A, int ml, int Kc,
                                            int kk, int lh, v16bf& ahi, v16bf& alo) {
    const float* ap = A + (size_t)ml * Kc + kk + (lh << 4);
    float4 a0 = *(const float4*)(ap + 0);
    float4 a1 = *(const float4*)(ap + 4);
    float4 a2 = *(const float4*)(ap + 8);
    float4 a3 = *(const float4*)(ap + 12);
    float av[16] = { a0.x, a0.y, a0.z, a0.w, a1.x, a1.y, a1.z, a1.w,
                     a2.x, a2.y, a2.z, a2.w, a3.x, a3.y, a3.z, a3.w };
    #pragma unroll
    for (int e = 0; e < 16; ++e) { __bf16 h, l; split_bf(av[e], h, l); ahi[e] = h; alo[e] = l; }
    __builtin_prefetch(ap + 32, 0, 0);   // global_prefetch_b8: next K-step panel
}

// B fragment (32x16; same contiguous k order as A): lane holds column n; k(e) = kk + (lh?16:0) + e
__device__ __forceinline__ void load_b_frag(const float* __restrict__ B, int n, int Nc,
                                            int kk, int lh, v16bf& bhi, v16bf& blo) {
    const float* bp = B + (size_t)(kk + (lh << 4)) * Nc + n;
    #pragma unroll
    for (int e = 0; e < 16; ++e) { __bf16 h, l; split_bf(bp[(size_t)e * Nc], h, l); bhi[e] = h; blo[e] = l; }
}

// B fragment from row-major [N,K] (transposed use, e.g. Gram A@A^T)
__device__ __forceinline__ void load_bt_frag(const float* __restrict__ B, int n, int Kc,
                                             int kk, int lh, v16bf& bhi, v16bf& blo) {
    const float* bp = B + (size_t)n * Kc + kk + (lh << 4);
    float4 b0 = *(const float4*)(bp + 0);
    float4 b1 = *(const float4*)(bp + 4);
    float4 b2 = *(const float4*)(bp + 8);
    float4 b3 = *(const float4*)(bp + 12);
    float bv[16] = { b0.x, b0.y, b0.z, b0.w, b1.x, b1.y, b1.z, b1.w,
                     b2.x, b2.y, b2.z, b2.w, b3.x, b3.y, b3.z, b3.w };
    #pragma unroll
    for (int e = 0; e < 16; ++e) { __bf16 h, l; split_bf(bv[e], h, l); bhi[e] = h; blo[e] = l; }
    __builtin_prefetch(bp + 32, 0, 0);
}

// bf16x3 compensated MAC: acc += A*B (small terms first)
__device__ __forceinline__ v8f wmma3(const v16bf& ahi, const v16bf& alo,
                                     const v16bf& bhi, const v16bf& blo, v8f acc) {
    acc = __builtin_amdgcn_wmma_f32_16x16x32_bf16(false, alo, false, bhi, (short)0, acc, false, false);
    acc = __builtin_amdgcn_wmma_f32_16x16x32_bf16(false, ahi, false, blo, (short)0, acc, false, false);
    acc = __builtin_amdgcn_wmma_f32_16x16x32_bf16(false, ahi, false, bhi, (short)0, acc, false, false);
    return acc;
}

// ---------------------------------------------------------------------------
// 1) Radius-graph bitmask
// ---------------------------------------------------------------------------
__global__ void build_mask_kernel(const float* __restrict__ X, unsigned* __restrict__ mask) {
    int i = blockIdx.x;
    int w = threadIdx.x;         // 128 words x 32 bits = 4096
    float xi = X[i * 3 + 0], yi = X[i * 3 + 1], zi = X[i * 3 + 2];
    unsigned m = 0;
    int base = w << 5;
    #pragma unroll 4
    for (int b = 0; b < 32; ++b) {
        int j = base + b;
        float dx = xi - X[j * 3 + 0];
        float dy = yi - X[j * 3 + 1];
        float dz = zi - X[j * 3 + 2];
        float d2 = dx * dx + dy * dy + dz * dz;
        if (d2 <= R2 && j != i) m |= (1u << b);
    }
    mask[i * 128 + w] = m;
}

// ---------------------------------------------------------------------------
// 2) Deterministic sparse aggregation: Agg[i,:] = sum_{j in N(i)} X[j,:]
// ---------------------------------------------------------------------------
__global__ void sparse_agg_kernel(const float* __restrict__ X, const unsigned* __restrict__ mask,
                                  float* __restrict__ Agg, int C) {
    __shared__ int list[NPTS];
    __shared__ int wcnt[128];
    __shared__ int woff[128];
    __shared__ int total;
    int i = blockIdx.x;
    int tid = threadIdx.x;
    const unsigned* mrow = mask + i * 128;
    if (tid < 128) wcnt[tid] = __popc(mrow[tid]);
    __syncthreads();
    if (tid == 0) {
        int s = 0;
        for (int w = 0; w < 128; ++w) { woff[w] = s; s += wcnt[w]; }
        total = s;
    }
    __syncthreads();
    if (tid < 128) {
        unsigned mw = mrow[tid];
        int o = woff[tid];
        int base = tid << 5;
        while (mw) {
            int b = __ffs(mw) - 1;
            list[o++] = base + b;
            mw &= (mw - 1);
        }
    }
    __syncthreads();
    int cnt = total;
    for (int c = tid; c < C; c += blockDim.x) {
        float s = 0.0f;
        for (int t = 0; t < cnt; ++t) s += X[(size_t)list[t] * C + c];
        Agg[(size_t)i * C + c] = s;
    }
}

// ---------------------------------------------------------------------------
// 3) Layer-0 GraphConv (K=3, tiny): h = x@Wr + agg@Wl + b
// ---------------------------------------------------------------------------
__global__ void gconv3_kernel(const float* __restrict__ X, const float* __restrict__ A3,
                              const float* __restrict__ Wr, const float* __restrict__ Wl,
                              const float* __restrict__ b, float* __restrict__ Out) {
    int idx = blockIdx.x * blockDim.x + threadIdx.x;   // NPTS*CCH
    int i = idx >> 8, c = idx & 255;
    float s = b[c];
    #pragma unroll
    for (int k = 0; k < 3; ++k)
        s += X[i * 3 + k] * Wr[k * CCH + c] + A3[i * 3 + k] * Wl[k * CCH + c];
    Out[idx] = s;
}

// ---------------------------------------------------------------------------
// 4a) Fused dual GEMM (GraphConv): Out = A1@B1 + A2@B2 + bias
//     One wave -> 16x64 output (4 D tiles), A fragments reused across tiles.
// ---------------------------------------------------------------------------
__global__ void gemm_dual_bf16x3_kernel(const float* __restrict__ A1, const float* __restrict__ B1,
                                        const float* __restrict__ A2, const float* __restrict__ B2,
                                        const float* __restrict__ bias, float* __restrict__ Out,
                                        int M, int Nc, int Kc) {
    int wid  = blockIdx.x * (blockDim.x >> 5) + (threadIdx.x >> 5);
    int lane = threadIdx.x & 31;
    int tilesN = Nc >> 6;                     // 64-wide wave tiles
    int tiles  = (M >> 4) * tilesN;
    if (wid >= tiles) return;                 // wave-uniform: EXEC stays all-1s
    int tm = wid / tilesN, tn = wid % tilesN;
    int lh = lane >> 4;
    int ml = (tm << 4) + (lane & 15);
    int nb = (tn << 6) + (lane & 15);

    v8f acc[4] = {};
    for (int kk = 0; kk < Kc; kk += 32) {
        v16bf ahi, alo;
        load_a_frag(A1, ml, Kc, kk, lh, ahi, alo);
        #pragma unroll
        for (int j = 0; j < 4; ++j) {
            v16bf bhi, blo;
            load_b_frag(B1, nb + (j << 4), Nc, kk, lh, bhi, blo);
            acc[j] = wmma3(ahi, alo, bhi, blo, acc[j]);
        }
        load_a_frag(A2, ml, Kc, kk, lh, ahi, alo);
        #pragma unroll
        for (int j = 0; j < 4; ++j) {
            v16bf bhi, blo;
            load_b_frag(B2, nb + (j << 4), Nc, kk, lh, bhi, blo);
            acc[j] = wmma3(ahi, alo, bhi, blo, acc[j]);
        }
    }
    #pragma unroll
    for (int j = 0; j < 4; ++j) {
        int n = nb + (j << 4);
        float bs = bias[n];
        #pragma unroll
        for (int v = 0; v < 8; ++v)
            Out[(size_t)((tm << 4) + (lh << 3) + v) * Nc + n] = acc[j][v] + bs;
    }
}

// ---------------------------------------------------------------------------
// 4b) Single GEMM: Out = A@B (+bias)(ReLU); BT: B row-major [N,K] (Gram)
// ---------------------------------------------------------------------------
template<bool BT, bool BIAS, bool RELU>
__global__ void gemm_bf16x3_kernel(const float* __restrict__ A, const float* __restrict__ B,
                                   const float* __restrict__ bias, float* __restrict__ Out,
                                   int M, int Nc, int Kc) {
    int wid  = blockIdx.x * (blockDim.x >> 5) + (threadIdx.x >> 5);
    int lane = threadIdx.x & 31;
    int tilesN = Nc >> 6;
    int tiles  = (M >> 4) * tilesN;
    if (wid >= tiles) return;
    int tm = wid / tilesN, tn = wid % tilesN;
    int lh = lane >> 4;
    int ml = (tm << 4) + (lane & 15);
    int nb = (tn << 6) + (lane & 15);

    v8f acc[4] = {};
    for (int kk = 0; kk < Kc; kk += 32) {
        v16bf ahi, alo;
        load_a_frag(A, ml, Kc, kk, lh, ahi, alo);
        #pragma unroll
        for (int j = 0; j < 4; ++j) {
            v16bf bhi, blo;
            if (BT) load_bt_frag(B, nb + (j << 4), Kc, kk, lh, bhi, blo);
            else    load_b_frag (B, nb + (j << 4), Nc, kk, lh, bhi, blo);
            acc[j] = wmma3(ahi, alo, bhi, blo, acc[j]);
        }
    }
    #pragma unroll
    for (int j = 0; j < 4; ++j) {
        int n = nb + (j << 4);
        float bs = BIAS ? bias[n] : 0.0f;
        #pragma unroll
        for (int v = 0; v < 8; ++v) {
            float r = acc[j][v] + bs;
            if (RELU) r = fmaxf(r, 0.0f);
            Out[(size_t)((tm << 4) + (lh << 3) + v) * Nc + n] = r;
        }
    }
}

// ---------------------------------------------------------------------------
// 5) Per-row top-16 (smallest d2, smallest-index tie-break) from Gram matrix
// ---------------------------------------------------------------------------
__global__ void topk_kernel(const float* __restrict__ G, int* __restrict__ idx) {
    __shared__ float d2s[NPTS];
    __shared__ float rv[256];
    __shared__ int   ri[256];
    int i = blockIdx.x, tid = threadIdx.x;
    float gii = G[(size_t)i * NPTS + i];
    for (int j = tid; j < NPTS; j += 256) {
        float v = gii + G[(size_t)j * NPTS + j] - 2.0f * G[(size_t)i * NPTS + j];
        d2s[j] = (j == i) ? 3.4e38f : v;
    }
    __syncthreads();
    for (int t = 0; t < KNN; ++t) {
        float bv = 3.4e38f; int bi = NPTS;
        for (int j = tid; j < NPTS; j += 256) {
            float v = d2s[j];
            if (v < bv || (v == bv && j < bi)) { bv = v; bi = j; }
        }
        rv[tid] = bv; ri[tid] = bi;
        __syncthreads();
        for (int s = 128; s > 0; s >>= 1) {
            if (tid < s) {
                float ov = rv[tid + s]; int oi = ri[tid + s];
                if (ov < rv[tid] || (ov == rv[tid] && oi < ri[tid])) { rv[tid] = ov; ri[tid] = oi; }
            }
            __syncthreads();
        }
        if (tid == 0) { idx[i * KNN + t] = ri[0]; d2s[ri[0]] = 3.4e38f; }
        __syncthreads();
    }
}

// ---------------------------------------------------------------------------
// 6) kNN aggregation: Agg[i,:] = sum_{t<16} H[idx[i][t],:]
// ---------------------------------------------------------------------------
__global__ void knn_agg_kernel(const float* __restrict__ H, const int* __restrict__ idx,
                               float* __restrict__ Agg) {
    int i = blockIdx.x, c = threadIdx.x;
    const int* ip = idx + i * KNN;
    float s = 0.0f;
    #pragma unroll
    for (int t = 0; t < KNN; ++t) s += H[(size_t)ip[t] * CCH + c];
    Agg[(size_t)i * CCH + c] = s;
}

// ---------------------------------------------------------------------------
// 7) Final tiny GEMM: [16384,64] @ [64,3] + b
// ---------------------------------------------------------------------------
__global__ void mlp_out3_kernel(const float* __restrict__ H, const float* __restrict__ W,
                                const float* __restrict__ b, float* __restrict__ Out) {
    int r = blockIdx.x * blockDim.x + threadIdx.x;
    const float* hp = H + (size_t)r * 64;
    float s0 = b[0], s1 = b[1], s2 = b[2];
    #pragma unroll 8
    for (int k = 0; k < 64; ++k) {
        float h = hp[k];
        s0 += h * W[k * 3 + 0];
        s1 += h * W[k * 3 + 1];
        s2 += h * W[k * 3 + 2];
    }
    Out[r * 3 + 0] = s0; Out[r * 3 + 1] = s1; Out[r * 3 + 2] = s2;
}

// ---------------------------------------------------------------------------
// Host launcher
// ---------------------------------------------------------------------------
extern "C" void kernel_launch(void* const* d_in, const int* in_sizes, int n_in,
                              void* d_out, int out_size, void* d_ws, size_t ws_size,
                              hipStream_t stream) {
    (void)in_sizes; (void)n_in; (void)out_size; (void)ws_size;

    const float* x       = (const float*)d_in[0];
    const float* w_root0 = (const float*)d_in[1];
    const float* w_rel0  = (const float*)d_in[2];
    const float* b0      = (const float*)d_in[3];
    const float* w_root1 = (const float*)d_in[4];
    const float* w_rel1  = (const float*)d_in[5];
    const float* b1      = (const float*)d_in[6];
    const float* w_root2 = (const float*)d_in[7];
    const float* w_rel2  = (const float*)d_in[8];
    const float* b2      = (const float*)d_in[9];
    const float* ns_wr   = (const float*)d_in[10];
    const float* ns_wl   = (const float*)d_in[11];
    const float* ns_b    = (const float*)d_in[12];
    const float* m_w0    = (const float*)d_in[13];
    const float* m_b0    = (const float*)d_in[14];
    const float* m_w1    = (const float*)d_in[15];
    const float* m_b1    = (const float*)d_in[16];
    const float* m_w2    = (const float*)d_in[17];
    const float* m_b2    = (const float*)d_in[18];
    float* out = (float*)d_out;

    // workspace layout (256B-aligned offsets), total ~116 MB
    char* ws = (char*)d_ws;
    size_t o = 0;
    unsigned* mask = (unsigned*)(ws + o); o += (size_t)NPTS * 128 * 4;          // 2 MB
    float* agg3 = (float*)(ws + o); o += ((size_t)NPTS * 3 * 4 + 255) & ~255ull;// 48 KB
    float* hA   = (float*)(ws + o); o += (size_t)NPTS * CCH * 4;                // 4 MB
    float* agg  = (float*)(ws + o); o += (size_t)NPTS * CCH * 4;                // 4 MB (reused)
    float* hB   = (float*)(ws + o); o += (size_t)NPTS * CCH * 4;                // 4 MB
    float* hC   = (float*)(ws + o); o += (size_t)NPTS * CCH * 4;                // 4 MB
    float* gram = (float*)(ws + o); o += (size_t)NPTS * NPTS * 4;               // 64 MB
    int*   kidx = (int*)  (ws + o); o += (size_t)NPTS * KNN * 4;                // 256 KB
    float* hN   = (float*)(ws + o); o += (size_t)NPTS * RUP * CCH * 4;          // 16 MB
    float* m1   = (float*)(ws + o); o += (size_t)NPTS * RUP * 128 * 4;          // 8 MB
    float* m2   = (float*)(ws + o); o += (size_t)NPTS * RUP * 64 * 4;           // 4 MB

    const int MROWS = NPTS * RUP;   // 16384

    // radius graph + layer 0
    build_mask_kernel<<<NPTS, 128, 0, stream>>>(x, mask);
    sparse_agg_kernel<<<NPTS, 256, 0, stream>>>(x, mask, agg3, 3);
    gconv3_kernel<<<NPTS, 256, 0, stream>>>(x, agg3, w_root0, w_rel0, b0, hA);

    // layer 1: hB = hA@Wroot1 + (adj@hA)@Wrel1 + b1   (fused dual GEMM)
    sparse_agg_kernel<<<NPTS, 256, 0, stream>>>(hA, mask, agg, CCH);
    { int t = (NPTS/16)*(CCH/64);
      gemm_dual_bf16x3_kernel<<<(t+3)/4,128,0,stream>>>(hA, w_root1, agg, w_rel1, b1, hB, NPTS, CCH, CCH); }

    // layer 2
    sparse_agg_kernel<<<NPTS, 256, 0, stream>>>(hB, mask, agg, CCH);
    { int t = (NPTS/16)*(CCH/64);
      gemm_dual_bf16x3_kernel<<<(t+3)/4,128,0,stream>>>(hB, w_root2, agg, w_rel2, b2, hC, NPTS, CCH, CCH); }

    // kNN graph in feature space (Gram via WMMA, B^T path)
    { int t = (NPTS/16)*(NPTS/64);
      gemm_bf16x3_kernel<true ,false,false><<<(t+3)/4,128,0,stream>>>(hC, hC, nullptr, gram, NPTS, NPTS, CCH); }
    topk_kernel<<<NPTS, 256, 0, stream>>>(gram, kidx);
    knn_agg_kernel<<<NPTS, 256, 0, stream>>>(hC, kidx, agg);

    // NodeShuffle GraphConv 256 -> 1024 ; [4096,1024] == [16384,256] row-major
    { int t = (NPTS/16)*((RUP*CCH)/64);
      gemm_dual_bf16x3_kernel<<<(t+3)/4,128,0,stream>>>(hC, ns_wr, agg, ns_wl, ns_b, hN, NPTS, RUP*CCH, CCH); }

    // reconstructor MLP
    { int t = (MROWS/16)*(128/64);
      gemm_bf16x3_kernel<false,true ,true ><<<(t+3)/4,128,0,stream>>>(hN, m_w0, m_b0, m1, MROWS, 128, CCH); }
    { int t = (MROWS/16)*(64/64);
      gemm_bf16x3_kernel<false,true ,true ><<<(t+3)/4,128,0,stream>>>(m1, m_w1, m_b1, m2, MROWS, 64, 128); }
    mlp_out3_kernel<<<MROWS/256, 256, 0, stream>>>(m2, m_w2, m_b2, out);
}
